// MultiheadAttention_29609504539205
// MI455X (gfx1250) — matrix operand order repaired and checked by reading, not compiled, not run on hardware
//
#include <hip/hip_runtime.h>

#define TGT 2048
#define BSZ 4
#define EMB 1024
#define NH  16
#define HD  64
#define MR  (TGT * BSZ)   // 8192 rows in [T,B,E] flattening

typedef unsigned short u16;
typedef __attribute__((ext_vector_type(16))) __bf16 v16bf;
typedef __attribute__((ext_vector_type(8)))  __bf16 v8bf;
typedef __attribute__((ext_vector_type(8)))  float  v8f;

union V16 { v16bf v; v8bf h[2]; };

static __device__ __forceinline__ u16 f2bf(float f) {
  union { float f; unsigned u; } x; x.f = f;
  unsigned r = x.u + 0x7FFFu + ((x.u >> 16) & 1u);  // round-to-nearest-even
  return (u16)(r >> 16);
}

// ---------------------------------------------------------------- f32 -> bf16
__global__ __launch_bounds__(256) void cvt_f32_bf16(const float* __restrict__ src,
                                                    u16* __restrict__ dst, long n) {
  long stride = (long)gridDim.x * blockDim.x;
  for (long i = (long)blockIdx.x * blockDim.x + threadIdx.x; i < n; i += stride)
    dst[i] = f2bf(src[i]);
}

// ---------------------------------------------------------------- GEMM 16x64 tile / wave
// C[m,n] = sum_k X[m,k] * W[n,k] + bias[n]      (y = x @ W^T + b)
// A (X): row-major MR x EMB bf16. B (W): row-major EMB x EMB bf16 (row n = output col n).
// mode 0: store bf16 head-major   dst[bh][t][d]   (Q, K)
// mode 1: store bf16 head-major-T dst[bh][d][t]   (V transposed for PV WMMA B operand)
// mode 2: store f32 row-major [m][n] into dst_f32 (final output projection)
__global__ __launch_bounds__(32) void gemm16x64(const u16* __restrict__ X,
                                                const u16* __restrict__ W,
                                                const float* __restrict__ bias,
                                                u16* __restrict__ dst_bf,
                                                float* __restrict__ dst_f32,
                                                int mode) {
  const int lane = threadIdx.x & 31;
  const int half = lane >> 4;          // 0: lanes 0-15, 1: lanes 16-31
  const int l16  = lane & 15;
  const int m0 = blockIdx.x * 16;
  const int n0 = blockIdx.y * 64;

  v8f acc0 = {}, acc1 = {}, acc2 = {}, acc3 = {};

  const u16* arow = X + (size_t)(m0 + l16) * EMB;                 // A: row m0+l16 over k
  const u16* b0 = W + (size_t)(n0 +  0 + l16) * EMB + half * 16;  // B: col n over k (row of W)
  const u16* b1 = W + (size_t)(n0 + 16 + l16) * EMB + half * 16;
  const u16* b2 = W + (size_t)(n0 + 32 + l16) * EMB + half * 16;
  const u16* b3 = W + (size_t)(n0 + 48 + l16) * EMB + half * 16;

#pragma unroll 2
  for (int k = 0; k < EMB; k += 32) {
    V16 a;                                           // A layout: 16 consecutive k per lane (k = half*16..+16)
    a.h[0] = *(const v8bf*)(arow + k + half * 16);
    a.h[1] = *(const v8bf*)(arow + k + half * 16 + 8);
    v16bf bb;
    bb = *(const v16bf*)(b0 + k);                    // B layout: 16 consecutive k per lane
    acc0 = __builtin_amdgcn_wmma_f32_16x16x32_bf16(false, a.v, false, bb, (short)0, acc0, false, false);
    bb = *(const v16bf*)(b1 + k);
    acc1 = __builtin_amdgcn_wmma_f32_16x16x32_bf16(false, a.v, false, bb, (short)0, acc1, false, false);
    bb = *(const v16bf*)(b2 + k);
    acc2 = __builtin_amdgcn_wmma_f32_16x16x32_bf16(false, a.v, false, bb, (short)0, acc2, false, false);
    bb = *(const v16bf*)(b3 + k);
    acc3 = __builtin_amdgcn_wmma_f32_16x16x32_bf16(false, a.v, false, bb, (short)0, acc3, false, false);
  }

  v8f accs[4] = {acc0, acc1, acc2, acc3};
#pragma unroll
  for (int j = 0; j < 4; ++j) {
    const int n  = n0 + j * 16 + l16;     // C/D layout: lane holds fixed col n
    const float bv = bias[n];
#pragma unroll
    for (int i = 0; i < 8; ++i) {
      const int m = m0 + i + half * 8;    // rows M = i (+8 for upper half)
      const float val = accs[j][i] + bv;
      if (mode == 2) {
        dst_f32[(size_t)m * EMB + n] = val;
      } else {
        const int t = m >> 2, b = m & 3;            // m = t*BSZ + b
        const int h = n >> 6, d = n & 63;           // n = h*HD + d
        const int bh = b * NH + h;
        size_t idx = (mode == 0)
            ? (((size_t)bh * TGT + t) * HD + d)     // [bh][t][d]
            : (((size_t)bh * HD + d) * TGT + t);    // [bh][d][t]
        dst_bf[idx] = f2bf(val);
      }
    }
  }
}

// ---------------------------------------------------------------- flash attention, 1 wave / (head, 16 q-rows)
__global__ __launch_bounds__(32) void attn_flash(const u16* __restrict__ Qh,  // [BH][T][HD]
                                                 const u16* __restrict__ Kh,  // [BH][S][HD]
                                                 const u16* __restrict__ Vt,  // [BH][HD][S]
                                                 u16* __restrict__ O) {       // [T][B][E] bf16
  __shared__ __align__(32) u16 pbuf[16 * 32];       // P tile bounce buffer (C-layout -> A-layout)

  const int lane = threadIdx.x & 31;
  const int half = lane >> 4;
  const int l16  = lane & 15;
  const int bh = blockIdx.x;                        // 0..63
  const int tt = blockIdx.y;                        // 0..127 (16-row q tile)

  const u16* qrow = Qh + ((size_t)bh * TGT + tt * 16 + l16) * HD;
  V16 qa0, qa1;                                     // Q as A operand, k-chunks d=[0,32) and [32,64)
  qa0.h[0] = *(const v8bf*)(qrow + half * 16);
  qa0.h[1] = *(const v8bf*)(qrow + half * 16 + 8);
  qa1.h[0] = *(const v8bf*)(qrow + 32 + half * 16);
  qa1.h[1] = *(const v8bf*)(qrow + 40 + half * 16);

  const u16* kbase = Kh + (size_t)bh * TGT * HD;
  const u16* vbase = Vt + (size_t)bh * HD * TGT;
  const float scale = 0.125f;                       // 1/sqrt(64)

  float mrow[8], lrow[8];
  v8f oacc[4] = {};
#pragma unroll
  for (int i = 0; i < 8; ++i) { mrow[i] = -3.0e38f; lrow[i] = 0.0f; }

#pragma unroll 1
  for (int s0 = 0; s0 < TGT; s0 += 32) {
    // ---- scores: two 16x16 tiles (cols s0..+15 and s0+16..+31), K-dim = d (64, two chunks)
    v8f c0 = {}, c1 = {};
    {
      const u16* kr = kbase + (size_t)(s0 + l16) * HD + half * 16;
      v16bf kb;
      kb = *(const v16bf*)(kr);
      c0 = __builtin_amdgcn_wmma_f32_16x16x32_bf16(false, qa0.v, false, kb, (short)0, c0, false, false);
      kb = *(const v16bf*)(kr + 32);
      c0 = __builtin_amdgcn_wmma_f32_16x16x32_bf16(false, qa1.v, false, kb, (short)0, c0, false, false);
      const u16* kr1 = kr + (size_t)16 * HD;
      kb = *(const v16bf*)(kr1);
      c1 = __builtin_amdgcn_wmma_f32_16x16x32_bf16(false, qa0.v, false, kb, (short)0, c1, false, false);
      kb = *(const v16bf*)(kr1 + 32);
      c1 = __builtin_amdgcn_wmma_f32_16x16x32_bf16(false, qa1.v, false, kb, (short)0, c1, false, false);
    }

    // ---- online softmax (row = i + 8*half; row spread over 16 lanes of one half)
    float tmax[8], alpha[8], tsum[8];
#pragma unroll
    for (int i = 0; i < 8; ++i) {
      c0[i] *= scale; c1[i] *= scale;
      tmax[i] = fmaxf(c0[i], c1[i]);
    }
#pragma unroll
    for (int i = 0; i < 8; ++i)
      for (int m = 8; m >= 1; m >>= 1)
        tmax[i] = fmaxf(tmax[i], __shfl_xor(tmax[i], m, 32));   // masks<16: stays in half
#pragma unroll
    for (int i = 0; i < 8; ++i) {
      const float mn = fmaxf(mrow[i], tmax[i]);
      alpha[i] = __expf(mrow[i] - mn);
      mrow[i] = mn;
      const float e0 = __expf(c0[i] - mn);
      const float e1 = __expf(c1[i] - mn);
      tsum[i] = e0 + e1;
      const int row = i + 8 * half;
      pbuf[row * 32 + l16]      = f2bf(e0);        // C-layout element -> row-major P tile
      pbuf[row * 32 + 16 + l16] = f2bf(e1);
    }
#pragma unroll
    for (int i = 0; i < 8; ++i)
      for (int m = 8; m >= 1; m >>= 1)
        tsum[i] += __shfl_xor(tsum[i], m, 32);
#pragma unroll
    for (int i = 0; i < 8; ++i) lrow[i] = lrow[i] * alpha[i] + tsum[i];
#pragma unroll
    for (int j = 0; j < 4; ++j)
#pragma unroll
      for (int i = 0; i < 8; ++i) oacc[j][i] *= alpha[i];

    __syncthreads();                                 // cross-lane LDS RAW
    V16 pA;                                          // P as A operand: row l16, k = s-offset
    pA.h[0] = *(const v8bf*)(pbuf + l16 * 32 + half * 16);
    pA.h[1] = *(const v8bf*)(pbuf + l16 * 32 + half * 16 + 8);
    __syncthreads();                                 // WAR before next iter's writes

    // ---- O += P @ V   (B operand from transposed V: contiguous s per lane)
#pragma unroll
    for (int j = 0; j < 4; ++j) {
      const u16* vr = vbase + (size_t)(j * 16 + l16) * TGT + s0 + half * 16;
      v16bf vb = *(const v16bf*)(vr);
      oacc[j] = __builtin_amdgcn_wmma_f32_16x16x32_bf16(false, pA.v, false, vb, (short)0, oacc[j], false, false);
    }
  }

  // ---- epilogue: normalize + scatter back to [T,B,E]
  const int b = bh >> 4, h = bh & 15;
#pragma unroll
  for (int j = 0; j < 4; ++j) {
    const int n = h * HD + j * 16 + l16;
#pragma unroll
    for (int i = 0; i < 8; ++i) {
      const int t = tt * 16 + i + 8 * half;
      const float val = oacc[j][i] / lrow[i];
      O[((size_t)t * BSZ + b) * EMB + n] = f2bf(val);
    }
  }
}

// ----------------------------------------------------------------
extern "C" void kernel_launch(void* const* d_in, const int* in_sizes, int n_in,
                              void* d_out, int out_size, void* d_ws, size_t ws_size,
                              hipStream_t stream) {
  const float* query = (const float*)d_in[0];
  const float* key   = (const float*)d_in[1];
  const float* value = (const float*)d_in[2];
  const float* Wq = (const float*)d_in[3];  const float* bq = (const float*)d_in[4];
  const float* Wk = (const float*)d_in[5];  const float* bk = (const float*)d_in[6];
  const float* Wv = (const float*)d_in[7];  const float* bv = (const float*)d_in[8];
  const float* Wo = (const float*)d_in[9];  const float* bo = (const float*)d_in[10];
  float* out = (float*)d_out;

  const size_t XN = (size_t)MR * EMB;     // 8M elems
  const size_t WN = (size_t)EMB * EMB;    // 1M elems
  u16* ws = (u16*)d_ws;
  u16* Xq = ws;           u16* Xk = Xq + XN;  u16* Xv = Xk + XN;
  u16* W_q = Xv + XN;     u16* W_k = W_q + WN; u16* W_v = W_k + WN; u16* W_o = W_v + WN;
  u16* Qh = W_o + WN;     u16* Kh = Qh + XN;  u16* Vt = Kh + XN;   u16* Oc = Vt + XN;

  // f32 -> bf16 staging
  cvt_f32_bf16<<<4096, 256, 0, stream>>>(query, Xq, (long)XN);
  cvt_f32_bf16<<<4096, 256, 0, stream>>>(key,   Xk, (long)XN);
  cvt_f32_bf16<<<4096, 256, 0, stream>>>(value, Xv, (long)XN);
  cvt_f32_bf16<<<2048, 256, 0, stream>>>(Wq, W_q, (long)WN);
  cvt_f32_bf16<<<2048, 256, 0, stream>>>(Wk, W_k, (long)WN);
  cvt_f32_bf16<<<2048, 256, 0, stream>>>(Wv, W_v, (long)WN);
  cvt_f32_bf16<<<2048, 256, 0, stream>>>(Wo, W_o, (long)WN);

  // Q/K/V projections (head-major; V transposed per head)
  dim3 g(MR / 16, EMB / 64);   // 512 x 16
  gemm16x64<<<g, 32, 0, stream>>>(Xq, W_q, bq, Qh, nullptr, 0);
  gemm16x64<<<g, 32, 0, stream>>>(Xk, W_k, bk, Kh, nullptr, 0);
  gemm16x64<<<g, 32, 0, stream>>>(Xv, W_v, bv, Vt, nullptr, 1);

  // flash attention: 64 heads x 128 q-tiles
  attn_flash<<<dim3(BSZ * NH, TGT / 16), 32, 0, stream>>>(Qh, Kh, Vt, Oc);

  // output projection -> f32
  gemm16x64<<<g, 32, 0, stream>>>(Oc, W_o, bo, nullptr, out, 2);
}
